// GAUEncoder_90572270338410
// MI455X (gfx1250) — matrix operand order repaired and checked by reading, not compiled, not run on hardware
//
#include <hip/hip_runtime.h>
#include <math.h>

// ---------------------------------------------------------------------------
// GAU encoder for MI455X (gfx1250, wave32).
// B=8, K=2048, D=512, E=1024, S=128.  M = B*K = 16384.
// GEMMs via v_wmma_f32_16x16x32_bf16.
// A-tiles: GLOBAL_LOAD_ASYNC_TO_LDS_B128 (ASYNCcnt, double buffered).
// B-tiles: TENSOR_LOAD_TO_LDS via TDM descriptor (TENSORcnt, double buffered).
// ---------------------------------------------------------------------------

#define Bdim 8
#define Kdim 2048
#define Ddim 512
#define Edim 1024
#define Sdim 128
#define Mdim (Bdim * Kdim)       // 16384
#define Fdim (2 * Edim + Sdim)   // 2176

typedef __attribute__((ext_vector_type(16))) __bf16 v16bf;
typedef __attribute__((ext_vector_type(8)))  float  v8f;
typedef __attribute__((ext_vector_type(4)))  int    v4i;
typedef __attribute__((ext_vector_type(8)))  int    v8i;

static __device__ inline v8f wmma_bf16(v16bf a, v16bf b, v8f c) {
  // 8 args: (neg_a, A, neg_b, B, c_mod, C, reuse_a, reuse_b)
  return __builtin_amdgcn_wmma_f32_16x16x32_bf16(false, a, false, b, (short)0, c,
                                                 false, false);
}

static __device__ inline v8f zero8() {
  v8f z;
#pragma unroll
  for (int i = 0; i < 8; ++i) z[i] = 0.0f;
  return z;
}

// ---------------------------------------------------------------------------
// Async global -> LDS copy of 16 bytes (GLOBAL_LOAD_ASYNC_TO_LDS_B128).
// Low 32 bits of an LDS-aperture flat pointer are the LDS byte address
// (ISA 10.2 aperture mapping).  Tracked by ASYNCcnt.
// ---------------------------------------------------------------------------
static __device__ inline void async_cp16(void* lds_dst, const void* gsrc) {
  const unsigned lds_off = (unsigned)(size_t)lds_dst;
  const unsigned long long ga = (unsigned long long)(size_t)gsrc;
  asm volatile("global_load_async_to_lds_b128 %0, %1, off"
               :
               : "v"(lds_off), "v"(ga)
               : "memory");
}

#define WAIT_ASYNC(n) asm volatile("s_wait_asynccnt " #n ::: "memory")

// ---------------------------------------------------------------------------
// TDM: load a 2D tile (tile_w bf16 per row, tile_h rows, row stride
// `stride` elements) from global into LDS, padding each 64B row group with
// 16B so the LDS layout matches the padded tile stride (LD = 40 bf16).
// D# per ISA 8.3/8.4: 2D form -> groups 0,1 only (VADDR2/3 omitted).
//   group0: count=1 | lds_addr | global_addr[56:0] | type=2
//   group1: wg_mask=0 (not in cluster), data_size=1 (2B),
//           pad_enable, pad_interval=3 (16 DW), pad_amount=3 (4 DW),
//           tensor_dim = tile_dim (no OOB), dim0_stride = row stride.
// Issued by one wave (EXEC ignored, SGPR descriptor); tracked by TENSORcnt.
// ---------------------------------------------------------------------------
static __device__ inline void tdm_load_2d(void* lds_dst, const void* gsrc,
                                          int tile_w, int tile_h, int stride) {
  const unsigned lds_addr = (unsigned)(size_t)lds_dst;
  const unsigned long long ga = (unsigned long long)(size_t)gsrc;
  v4i g0;
  g0[0] = 1;                                           // count=1 (valid, user)
  g0[1] = (int)lds_addr;                               // lds_addr[31:0]
  g0[2] = (int)(unsigned)(ga & 0xFFFFFFFFu);           // global_addr[31:0]
  g0[3] = (int)((((unsigned)(ga >> 32)) & 0x01FFFFFFu) // global_addr[56:32]
                | 0x80000000u);                        // type=2 ("image")
  v8i g1;
  g1[0] = (1 << 16) | (1 << 20) | (3 << 22) | (3 << 25); // 2B, pad 16DW -> +4DW
  g1[1] = (tile_w & 0xFFFF) << 16;                     // tensor_dim0[15:0]
  g1[2] = ((tile_w >> 16) & 0xFFFF) | ((tile_h & 0xFFFF) << 16); // d0 hi | d1 lo
  g1[3] = ((tile_h >> 16) & 0xFFFF) | ((tile_w & 0xFFFF) << 16); // d1 hi | tile_dim0
  g1[4] = (tile_h & 0xFFFF);                           // tile_dim1 | tile_dim2=0
  g1[5] = stride;                                      // tensor_dim0_stride[31:0]
  g1[6] = 0;                                           // stride hi | dim1_stride lo
  g1[7] = 0;
  asm volatile("tensor_load_to_lds %0, %1" : : "s"(g0), "s"(g1) : "memory");
}

// A fragment: 16x32 bf16, row-major source (stride ld elements).
static __device__ inline v16bf load_a_frag(const __bf16* base, int ld, int lane) {
  const int half = lane >> 4;
  const int r = lane & 15;
  const __bf16* p = base + r * ld + half * 8;
  union { uint4 u[2]; v16bf v; } cvt;
  cvt.u[0] = *(const uint4*)(p);
  cvt.u[1] = *(const uint4*)(p + 16);
  return cvt.v;
}

// B fragment: 32x16 bf16 from an N-major LDS tile Bt[n][k], stride ld.
static __device__ inline v16bf load_b_frag(const __bf16* base, int ld, int lane) {
  const int half = lane >> 4;
  const int c = lane & 15;
  const __bf16* p = base + c * ld + half * 8;
  union { uint4 u[2]; v16bf v; } cvt;
  cvt.u[0] = *(const uint4*)(p);
  cvt.u[1] = *(const uint4*)(p + 16);
  return cvt.v;
}

// ---------------------------------------------------------------------------
// Kernel 1: convert weights fp32 -> bf16 (both already N-major).
// ---------------------------------------------------------------------------
__global__ __launch_bounds__(256) void k_convert_weights(
    const float* __restrict__ uv_w, const float* __restrict__ o_w,
    __bf16* __restrict__ uvwB, __bf16* __restrict__ owB) {
  const int n_uv = Fdim * Ddim;
  const int n_o  = Ddim * Edim;
  int i = blockIdx.x * 256 + threadIdx.x;
  if (i < n_uv) {
    uvwB[i] = (__bf16)uv_w[i];
  } else if (i < n_uv + n_o) {
    owB[i - n_uv] = (__bf16)o_w[i - n_uv];
  }
}

// ---------------------------------------------------------------------------
// Kernel 2: xn = x / max(||x||/sqrt(D), eps) * g, emit bf16.
// ---------------------------------------------------------------------------
__global__ __launch_bounds__(256) void k_rmsnorm(
    const float* __restrict__ x, const float* __restrict__ g,
    __bf16* __restrict__ xnb) {
  const int row  = blockIdx.x * 8 + (threadIdx.x >> 5);
  const int lane = threadIdx.x & 31;
  const float* xr = x + row * Ddim;
  float v[16];
  float ss = 0.0f;
#pragma unroll
  for (int i = 0; i < 16; ++i) {
    v[i] = xr[lane + i * 32];
    ss += v[i] * v[i];
  }
#pragma unroll
  for (int off = 16; off > 0; off >>= 1) ss += __shfl_xor(ss, off, 32);
  const float norm  = __builtin_sqrtf(ss * (1.0f / (float)Ddim));
  const float scale = g[0] / fmaxf(norm, 1e-5f);
  __bf16* o = xnb + row * Ddim;
#pragma unroll
  for (int i = 0; i < 16; ++i) o[lane + i * 32] = (__bf16)(v[i] * scale);
}

// ---------------------------------------------------------------------------
// Kernel 3: uv = silu(xn @ uv_w^T), 16384 x 2176, Kdim=512.
// A-tiles via ASYNCcnt, B-tiles via TDM, double buffered.
// ---------------------------------------------------------------------------
__global__ __launch_bounds__(256) void k_uv_gemm(
    const __bf16* __restrict__ xnb, const __bf16* __restrict__ uvwB,
    const float* __restrict__ gamma, const float* __restrict__ beta,
    __bf16* __restrict__ ub, __bf16* __restrict__ vT,
    __bf16* __restrict__ qb, __bf16* __restrict__ kb) {
  constexpr int LD = 40;  // 32 + 8 pad == TDM pad (16 DW + 4 DW)
  __shared__ __bf16 As[2][128 * LD];   // m x k
  __shared__ __bf16 Bt[2][128 * LD];   // n x k (N-major)
  const int ntile = blockIdx.x;        // 0..16
  const int mtile = blockIdx.y;        // 0..127
  const int tid = threadIdx.x, lane = tid & 31, wave = tid >> 5;
  const int wr = (wave >> 1) * 32;
  const int wc = (wave & 1) * 64;
  const int mbase = mtile * 128;
  const int nbase = ntile * 128;

  v8f acc[2][4];
#pragma unroll
  for (int i = 0; i < 2; ++i)
#pragma unroll
    for (int j = 0; j < 4; ++j) acc[i][j] = zero8();

  // preload tile 0: A async (2 ops/wave), B via TDM (wave 0 only)
#pragma unroll
  for (int v = tid; v < 512; v += 256) {
    const int r = v >> 2, c8 = (v & 3) * 8;
    async_cp16(&As[0][r * LD + c8], &xnb[(mbase + r) * Ddim + c8]);
  }
  if (wave == 0) tdm_load_2d(&Bt[0][0], &uvwB[nbase * Ddim], 32, 128, Ddim);

  const int NT = Ddim / 32;  // 16
  for (int t = 0; t < NT; ++t) {
    const int cur = t & 1;
    if (t + 1 < NT) {
      const int kk = (t + 1) * 32;
      const int nxt = cur ^ 1;
#pragma unroll
      for (int v = tid; v < 512; v += 256) {
        const int r = v >> 2, c8 = (v & 3) * 8;
        async_cp16(&As[nxt][r * LD + c8], &xnb[(mbase + r) * Ddim + kk + c8]);
      }
      if (wave == 0)
        tdm_load_2d(&Bt[nxt][0], &uvwB[nbase * Ddim + kk], 32, 128, Ddim);
      WAIT_ASYNC(2);  // retire A tile t, keep tile t+1 in flight
      if (wave == 0) __builtin_amdgcn_s_wait_tensorcnt(1);  // retire B tile t
    } else {
      WAIT_ASYNC(0);
      if (wave == 0) __builtin_amdgcn_s_wait_tensorcnt(0);
    }
    __syncthreads();
    const v16bf a0 = load_a_frag(&As[cur][(wr + 0) * LD], LD, lane);
    const v16bf a1 = load_a_frag(&As[cur][(wr + 16) * LD], LD, lane);
#pragma unroll
    for (int j = 0; j < 4; ++j) {
      const v16bf b = load_b_frag(&Bt[cur][(wc + j * 16) * LD], LD, lane);
      acc[0][j] = wmma_bf16(a0, b, acc[0][j]);
      acc[1][j] = wmma_bf16(a1, b, acc[1][j]);
    }
    __syncthreads();  // done reading buffer `cur` before it is refilled
  }

  const int half = lane >> 4, nl = lane & 15;
#pragma unroll
  for (int fi = 0; fi < 2; ++fi) {
#pragma unroll
    for (int fj = 0; fj < 4; ++fj) {
#pragma unroll
      for (int e = 0; e < 8; ++e) {
        const int m = mbase + wr + fi * 16 + e + half * 8;
        const int c = nbase + wc + fj * 16 + nl;
        float s = acc[fi][fj][e];
        s = s / (1.0f + __expf(-s));  // silu
        if (ntile < 8) {
          ub[m * Edim + c] = (__bf16)s;
        } else if (ntile < 16) {
          const int ecol = c - Edim;
          const int b = m >> 11, key = m & (Kdim - 1);
          vT[(b * Edim + ecol) * Kdim + key] = (__bf16)s;
        } else {
          const int si = c - 2 * Edim;  // 0..127
          const float qv = s * gamma[si] + beta[si];
          const float kv = s * gamma[Sdim + si] + beta[Sdim + si];
          qb[m * Sdim + si] = (__bf16)qv;
          kb[m * Sdim + si] = (__bf16)kv;
        }
      }
    }
  }
}

// ---------------------------------------------------------------------------
// Kernel 4: fused attention (async-to-LDS staging + L2 prefetch).
// Block = (b, 64 q-rows, 128 e-cols). Loop over key tiles of 64:
//   scores = q @ k^T / sqrt(S)  -> relu^2 -> bf16 LDS tile -> agg += @ v.
// ---------------------------------------------------------------------------
__global__ __launch_bounds__(256) void k_attn(
    const __bf16* __restrict__ qb, const __bf16* __restrict__ kb,
    const __bf16* __restrict__ vT, const __bf16* __restrict__ ub,
    __bf16* __restrict__ aggb) {
  constexpr int LQ = 136, LK = 136, LV = 72, LC = 72;
  __shared__ __bf16 Qs[64 * LQ];    // q rows   x S     (A-major)
  __shared__ __bf16 Ks[64 * LK];    // key rows x S     (N-major for scores B)
  __shared__ __bf16 Vs[128 * LV];   // e rows   x key   (N-major for agg B)
  __shared__ __bf16 Cs[64 * LC];    // q rows   x key   (A-major kernel tile)
  const int etile = blockIdx.x;     // 0..7
  const int qtile = blockIdx.y;     // 0..31
  const int b     = blockIdx.z;     // 0..7
  const int tid = threadIdx.x, lane = tid & 31, wave = tid >> 5;
  const int wrow = (wave >> 2) * 32;
  const int scol = (wave & 3) * 16;
  const int acol = (wave & 3) * 32;
  const int half = lane >> 4, nl = lane & 15;
  const float inv_sqrt_s = 0.08838834764831845f;  // 1/sqrt(128)

  const int qrow0 = b * Kdim + qtile * 64;
#pragma unroll
  for (int v = tid; v < 1024; v += 256) {  // 64x128 Q tile (async)
    const int r = v >> 4, c8 = (v & 15) * 8;
    async_cp16(&Qs[r * LQ + c8], &qb[(qrow0 + r) * Sdim + c8]);
  }

  v8f agg[2][2];
#pragma unroll
  for (int i = 0; i < 2; ++i)
#pragma unroll
    for (int j = 0; j < 2; ++j) agg[i][j] = zero8();

  for (int kt = 0; kt < Kdim; kt += 64) {
    __syncthreads();  // all waves done reading Ks/Vs before refill
#pragma unroll
    for (int v = tid; v < 1024; v += 256) {  // 64 keys x 128 s
      const int r = v >> 4, c8 = (v & 15) * 8;
      async_cp16(&Ks[r * LK + c8], &kb[(b * Kdim + kt + r) * Sdim + c8]);
    }
#pragma unroll
    for (int v = tid; v < 1024; v += 256) {  // 128 e x 64 keys
      const int r = v >> 3, c8 = (v & 7) * 8;
      async_cp16(&Vs[r * LV + c8],
                 &vT[(b * Edim + etile * 128 + r) * Kdim + kt + c8]);
    }
    if (kt + 64 < Kdim) {  // warm L2 for the next key tile
      __builtin_prefetch(&kb[(b * Kdim + kt + 64 + (tid >> 2)) * Sdim], 0, 0);
    }
    WAIT_ASYNC(0);
    __syncthreads();

    // scores: rows [wrow,wrow+32) x cols [scol,scol+16), Kdim = S = 128
    v8f sacc[2];
    sacc[0] = zero8();
    sacc[1] = zero8();
#pragma unroll
    for (int ks = 0; ks < Sdim; ks += 32) {
      const v16bf a0 = load_a_frag(&Qs[(wrow + 0) * LQ + ks], LQ, lane);
      const v16bf a1 = load_a_frag(&Qs[(wrow + 16) * LQ + ks], LQ, lane);
      const v16bf bb = load_b_frag(&Ks[scol * LK + ks], LK, lane);
      sacc[0] = wmma_bf16(a0, bb, sacc[0]);
      sacc[1] = wmma_bf16(a1, bb, sacc[1]);
    }
    // relu^2 -> bf16 A-matrix tile for the agg GEMM
#pragma unroll
    for (int fi = 0; fi < 2; ++fi) {
#pragma unroll
      for (int e = 0; e < 8; ++e) {
        const int r = wrow + fi * 16 + e + half * 8;
        const int c = scol + nl;
        float sv = sacc[fi][e] * inv_sqrt_s;
        sv = (sv > 0.0f) ? sv * sv : 0.0f;
        Cs[r * LC + c] = (__bf16)sv;
      }
    }
    __syncthreads();

    // agg += kernel @ v : rows [wrow,+32) x cols [acol,+32), Kdim = 64
#pragma unroll
    for (int ks = 0; ks < 64; ks += 32) {
      const v16bf a0 = load_a_frag(&Cs[(wrow + 0) * LC + ks], LC, lane);
      const v16bf a1 = load_a_frag(&Cs[(wrow + 16) * LC + ks], LC, lane);
#pragma unroll
      for (int j = 0; j < 2; ++j) {
        const v16bf bb = load_b_frag(&Vs[(acol + j * 16) * LV + ks], LV, lane);
        agg[0][j] = wmma_bf16(a0, bb, agg[0][j]);
        agg[1][j] = wmma_bf16(a1, bb, agg[1][j]);
      }
    }
  }

  // epilogue: multiply by u, store bf16 agg
#pragma unroll
  for (int fi = 0; fi < 2; ++fi) {
#pragma unroll
    for (int fj = 0; fj < 2; ++fj) {
#pragma unroll
      for (int e = 0; e < 8; ++e) {
        const int m = qrow0 + wrow + fi * 16 + e + half * 8;
        const int ecol = etile * 128 + acol + fj * 16 + nl;
        const float uval = (float)ub[m * Edim + ecol];
        aggb[m * Edim + ecol] = (__bf16)(agg[fi][fj][e] * uval);
      }
    }
  }
}

// ---------------------------------------------------------------------------
// Kernel 5: out = x*res_scale + agg @ o_w^T.  16384 x 512, Kdim = 1024.
// Same async-A / TDM-B double-buffered pipeline.
// ---------------------------------------------------------------------------
__global__ __launch_bounds__(256) void k_out_gemm(
    const __bf16* __restrict__ aggb, const __bf16* __restrict__ owB,
    const float* __restrict__ x, const float* __restrict__ res_scale,
    float* __restrict__ out) {
  constexpr int LD = 40;
  __shared__ __bf16 As[2][128 * LD];
  __shared__ __bf16 Bt[2][128 * LD];
  const int ntile = blockIdx.x;  // 0..3
  const int mtile = blockIdx.y;  // 0..127
  const int tid = threadIdx.x, lane = tid & 31, wave = tid >> 5;
  const int wr = (wave >> 1) * 32;
  const int wc = (wave & 1) * 64;
  const int mbase = mtile * 128;
  const int nbase = ntile * 128;

  v8f acc[2][4];
#pragma unroll
  for (int i = 0; i < 2; ++i)
#pragma unroll
    for (int j = 0; j < 4; ++j) acc[i][j] = zero8();

#pragma unroll
  for (int v = tid; v < 512; v += 256) {
    const int r = v >> 2, c8 = (v & 3) * 8;
    async_cp16(&As[0][r * LD + c8], &aggb[(mbase + r) * Edim + c8]);
  }
  if (wave == 0) tdm_load_2d(&Bt[0][0], &owB[nbase * Edim], 32, 128, Edim);

  const int NT = Edim / 32;  // 32
  for (int t = 0; t < NT; ++t) {
    const int cur = t & 1;
    if (t + 1 < NT) {
      const int kk = (t + 1) * 32;
      const int nxt = cur ^ 1;
#pragma unroll
      for (int v = tid; v < 512; v += 256) {
        const int r = v >> 2, c8 = (v & 3) * 8;
        async_cp16(&As[nxt][r * LD + c8], &aggb[(mbase + r) * Edim + kk + c8]);
      }
      if (wave == 0)
        tdm_load_2d(&Bt[nxt][0], &owB[nbase * Edim + kk], 32, 128, Edim);
      WAIT_ASYNC(2);
      if (wave == 0) __builtin_amdgcn_s_wait_tensorcnt(1);
    } else {
      WAIT_ASYNC(0);
      if (wave == 0) __builtin_amdgcn_s_wait_tensorcnt(0);
    }
    __syncthreads();
    const v16bf a0 = load_a_frag(&As[cur][(wr + 0) * LD], LD, lane);
    const v16bf a1 = load_a_frag(&As[cur][(wr + 16) * LD], LD, lane);
#pragma unroll
    for (int j = 0; j < 4; ++j) {
      const v16bf b = load_b_frag(&Bt[cur][(wc + j * 16) * LD], LD, lane);
      acc[0][j] = wmma_bf16(a0, b, acc[0][j]);
      acc[1][j] = wmma_bf16(a1, b, acc[1][j]);
    }
    __syncthreads();
  }

  const int half = lane >> 4, nl = lane & 15;
#pragma unroll
  for (int fi = 0; fi < 2; ++fi) {
#pragma unroll
    for (int fj = 0; fj < 4; ++fj) {
#pragma unroll
      for (int e = 0; e < 8; ++e) {
        const int m = mbase + wr + fi * 16 + e + half * 8;
        const int n = nbase + wc + fj * 16 + nl;
        out[m * Ddim + n] = x[m * Ddim + n] * res_scale[n] + acc[fi][fj][e];
      }
    }
  }
}

// ---------------------------------------------------------------------------
// Launch
// ---------------------------------------------------------------------------
extern "C" void kernel_launch(void* const* d_in, const int* in_sizes, int n_in,
                              void* d_out, int out_size, void* d_ws,
                              size_t ws_size, hipStream_t stream) {
  const float* x     = (const float*)d_in[0];  // (B,K,D)
  const float* uv_w  = (const float*)d_in[1];  // (2E+S, D)
  const float* o_w   = (const float*)d_in[2];  // (D, E)
  const float* gamma = (const float*)d_in[3];  // (2, S)
  const float* beta  = (const float*)d_in[4];  // (2, S)
  const float* g     = (const float*)d_in[5];  // (1,)
  const float* res   = (const float*)d_in[6];  // (D,)
  float* out = (float*)d_out;

  char* ws = (char*)d_ws;
  __bf16* xnb  = (__bf16*)ws;  ws += (size_t)Mdim * Ddim * 2;   // 16 MB
  __bf16* uvwB = (__bf16*)ws;  ws += (size_t)Fdim * Ddim * 2;   // 2.2 MB
  __bf16* owB  = (__bf16*)ws;  ws += (size_t)Ddim * Edim * 2;   // 1 MB
  __bf16* ub   = (__bf16*)ws;  ws += (size_t)Mdim * Edim * 2;   // 32 MB
  __bf16* vT   = (__bf16*)ws;  ws += (size_t)Mdim * Edim * 2;   // 32 MB
  __bf16* qb   = (__bf16*)ws;  ws += (size_t)Mdim * Sdim * 2;   // 4 MB
  __bf16* kb   = (__bf16*)ws;  ws += (size_t)Mdim * Sdim * 2;   // 4 MB
  __bf16* aggb = (__bf16*)ws;  ws += (size_t)Mdim * Edim * 2;   // 32 MB

  {
    const int total = Fdim * Ddim + Ddim * Edim;
    k_convert_weights<<<(total + 255) / 256, 256, 0, stream>>>(uv_w, o_w, uvwB,
                                                               owB);
  }
  k_rmsnorm<<<Mdim / 8, 256, 0, stream>>>(x, g, xnb);
  {
    dim3 grid(Fdim / 128, Mdim / 128);  // (17, 128)
    k_uv_gemm<<<grid, 256, 0, stream>>>(xnb, uvwB, gamma, beta, ub, vT, qb, kb);
  }
  {
    dim3 grid(Edim / 128, Kdim / 64, Bdim);  // (8, 32, 8)
    k_attn<<<grid, 256, 0, stream>>>(qb, kb, vT, ub, aggb);
  }
  {
    dim3 grid(Ddim / 128, Mdim / 128);  // (4, 128)
    k_out_gemm<<<grid, 256, 0, stream>>>(aggb, owB, x, res, out);
  }
}
